// GAT_21526376088369
// MI455X (gfx1250) — matrix operand order repaired and checked by reading, not compiled, not run on hardware
//
#include <hip/hip_runtime.h>
#include <math.h>

// ---------------------------------------------------------------------------
// GAT forward on gfx1250 (MI455X): bf16 WMMA GEMMs + flash-style masked
// softmax attention fused per 16-row tile. Wave32, 16x16x32 bf16 WMMA.
// Both hot loops are software-pipelined one tile ahead.
// ---------------------------------------------------------------------------

typedef __bf16 bf16_t;
typedef __attribute__((ext_vector_type(16))) __bf16 v16bf;
typedef __attribute__((ext_vector_type(8)))  __bf16 v8bf;
typedef __attribute__((ext_vector_type(8)))  float  v8f;

constexpr int kB = 8;      // batch
constexpr int kN = 1024;   // nodes
constexpr int kNT = kN / 16;
constexpr float kAlpha = 0.2f;
constexpr float kNeg = -9e15f;

__device__ __forceinline__ v8f wmma_bf16(v16bf a, v16bf b, v8f c) {
  // D = A(16x32 bf16) * B(32x16 bf16) + C(16x16 f32)
  return __builtin_amdgcn_wmma_f32_16x16x32_bf16(
      /*neg_a=*/false, a, /*neg_b=*/false, b,
      /*c_mod=*/(short)0, c, /*reuse_a=*/false, /*reuse_b=*/false);
}

__device__ __forceinline__ float elu1(float x) {
  return x > 0.0f ? x : (__expf(x) - 1.0f);
}

// ---------------------------------------------------------------------------
// f32 -> bf16 elementwise conversion (for weight matrices, one-time)
// ---------------------------------------------------------------------------
__global__ void cvt_bf16_kernel(const float* __restrict__ in,
                                bf16_t* __restrict__ out, int total) {
  const int tid = blockIdx.x * blockDim.x + threadIdx.x;
  if (tid >= total) return;
  out[tid] = (bf16_t)in[tid];
}

// ---------------------------------------------------------------------------
// GEMM: Wh[b,hh,n,O] (bf16) = A[b,n,Kdim] @ Wbf[hh,Kdim,O] (bf16 weights)
// One wave computes a 16-row x 64-col tile (4 WMMA col tiles).
// K loop by 32, software-pipelined: next tile's loads issue before this
// tile's WMMAs; the A conversion (the only forced load-wait) lands after.
// ---------------------------------------------------------------------------
template <bool A_BF16>
__global__ void __launch_bounds__(128)
gemm_wh_kernel(const void* __restrict__ Ap, const bf16_t* __restrict__ Wbf,
               bf16_t* __restrict__ Wh, int Kdim, int O, int HH, int colChunks,
               int totalWaves) {
  const int wave = blockIdx.x * (blockDim.x >> 5) + (threadIdx.x >> 5);
  if (wave >= totalWaves) return;
  const int lane = threadIdx.x & 31;
  const int half = lane >> 4;
  const int mrow = lane & 15;

  int tmp = wave;
  const int cc = tmp % colChunks; tmp /= colChunks;
  const int rt = tmp % kNT;       tmp /= kNT;
  const int hh = tmp % HH;        tmp /= HH;
  const int b  = tmp;

  const int row0 = rt * 16;
  const int colbase = cc * 64;
  const int i = row0 + mrow;

  const v8f vzero = {0.f, 0.f, 0.f, 0.f, 0.f, 0.f, 0.f, 0.f};
  v8f acc0 = vzero, acc1 = vzero, acc2 = vzero, acc3 = vzero;

  const float*  Af = (const float*)Ap;
  const bf16_t* Ab = (const bf16_t*)Ap;
  const size_t arowBase = ((size_t)b * kN + i) * (size_t)Kdim;

  // raw (unconverted) A staging registers
  float4 ra0, ra1, ra2, ra3;  // f32 path
  v8bf   rb0, rb1;            // bf16 path

  auto issueA = [&](int k0) {
    if (A_BF16) {
      const bf16_t* ar = Ab + arowBase + k0 + 8 * half;
      rb0 = *(const v8bf*)(ar);
      rb1 = *(const v8bf*)(ar + 16);
    } else {
      const float* ar = Af + arowBase + k0 + 8 * half;
      ra0 = *(const float4*)(ar);
      ra1 = *(const float4*)(ar + 4);
      ra2 = *(const float4*)(ar + 16);
      ra3 = *(const float4*)(ar + 20);
    }
  };
  auto cvtA = [&]() -> v16bf {
    v16bf af;
    if (A_BF16) {
#pragma unroll
      for (int t = 0; t < 8; ++t) { af[t] = rb0[t]; af[t + 8] = rb1[t]; }
    } else {
      const float fa[16] = {ra0.x, ra0.y, ra0.z, ra0.w, ra1.x, ra1.y, ra1.z,
                            ra1.w, ra2.x, ra2.y, ra2.z, ra2.w, ra3.x, ra3.y,
                            ra3.z, ra3.w};
#pragma unroll
      for (int t = 0; t < 16; ++t) af[t] = (bf16_t)fa[t];
    }
    return af;
  };
  auto issueB = [&](int k0, v16bf& r0, v16bf& r1, v16bf& r2, v16bf& r3) {
    const bf16_t* wcol =
        Wbf + ((size_t)hh * Kdim + k0) * (size_t)O + colbase + (lane & 15);
#pragma unroll
    for (int s = 0; s < 16; ++s) {
      const int kk = (s < 8) ? (8 * half + s) : (16 + 8 * half + (s - 8));
      const bf16_t* wp = wcol + (size_t)kk * O;
      r0[s] = wp[0]; r1[s] = wp[16]; r2[s] = wp[32]; r3[s] = wp[48];
    }
  };

  // ---- prologue: tile k0 = 0
  v16bf b0, b1, b2, b3;
  issueA(0);
  issueB(0, b0, b1, b2, b3);
  v16bf af = cvtA();

#pragma unroll 2
  for (int k0 = 0; k0 + 32 < Kdim; k0 += 32) {
    // issue next tile's loads before consuming the current one
    v16bf c0, c1, c2, c3;
    issueB(k0 + 32, c0, c1, c2, c3);
    issueA(k0 + 32);

    acc0 = wmma_bf16(af, b0, acc0);
    acc1 = wmma_bf16(af, b1, acc1);
    acc2 = wmma_bf16(af, b2, acc2);
    acc3 = wmma_bf16(af, b3, acc3);

    af = cvtA();  // waits on next-A loads (after WMMAs are in flight)
    b0 = c0; b1 = c1; b2 = c2; b3 = c3;
  }

  // ---- peeled last tile
  acc0 = wmma_bf16(af, b0, acc0);
  acc1 = wmma_bf16(af, b1, acc1);
  acc2 = wmma_bf16(af, b2, acc2);
  acc3 = wmma_bf16(af, b3, acc3);

  // ---- store: C element t of lane -> row (t + 8*half), col (c*16 + mrow)
#pragma unroll
  for (int t = 0; t < 8; ++t) {
    const int r = t + 8 * half;
    const size_t base =
        (((size_t)b * HH + hh) * kN + row0 + r) * (size_t)O + colbase + mrow;
    Wh[base +  0] = (bf16_t)acc0[t];
    Wh[base + 16] = (bf16_t)acc1[t];
    Wh[base + 32] = (bf16_t)acc2[t];
    Wh[base + 48] = (bf16_t)acc3[t];
  }
}

// ---------------------------------------------------------------------------
// e_src / e_dst: per-row dot of Wh with attention vector halves
// ---------------------------------------------------------------------------
__global__ void edot_kernel(const bf16_t* __restrict__ Wh,
                            const float* __restrict__ a,
                            float* __restrict__ es, float* __restrict__ ed,
                            int HH, int O, int total) {
  const int tid = blockIdx.x * blockDim.x + threadIdx.x;
  if (tid >= total) return;
  const int hh = (tid / kN) % HH;
  const float* av = a + (size_t)hh * 2 * O;
  const bf16_t* row = Wh + (size_t)tid * O;
  float s0 = 0.f, s1 = 0.f;
  for (int k = 0; k < O; ++k) {
    const float w = (float)row[k];
    s0 += w * av[k];
    s1 += w * av[O + k];
  }
  es[tid] = s0;
  ed[tid] = s1;
}

// ---------------------------------------------------------------------------
// Flash-style masked softmax attention + aggregation.
// One wave: 16 rows x 64 output cols; streams neighbors j in chunks of 32.
// Adjacency/e_dst loads are pipelined one tile ahead; logits are built
// directly in the WMMA A-fragment layout; probabilities multiply against
// Wh B-fragments with v_wmma_f32_16x16x32_bf16.
// ---------------------------------------------------------------------------
template <bool OUT_BF16>
__global__ void __launch_bounds__(128)
flash_gat_kernel(const float* __restrict__ es, const float* __restrict__ ed,
                 const bf16_t* __restrict__ Wh, const int* __restrict__ adjs,
                 void* __restrict__ outp, int HH, int O, int colChunks,
                 int totalWaves) {
  const int wave = blockIdx.x * (blockDim.x >> 5) + (threadIdx.x >> 5);
  if (wave >= totalWaves) return;
  const int lane = threadIdx.x & 31;
  const int half = lane >> 4;
  const int mrow = lane & 15;

  int tmp = wave;
  const int cc = tmp % colChunks; tmp /= colChunks;
  const int rt = tmp % kNT;       tmp /= kNT;
  const int hh = tmp % HH;        tmp /= HH;
  const int b  = tmp;

  const int row0 = rt * 16;
  const int colbase = cc * 64;
  const int i = row0 + mrow;

  const size_t bhN = ((size_t)b * HH + hh) * (size_t)kN;
  const float eSrc = es[bhN + i];                    // lane's row logit bias
  const float* edp = ed + bhN;
  const int* adjp = adjs + ((size_t)b * kN + i) * (size_t)kN;
  const bf16_t* whp = Wh + bhN * (size_t)O + colbase;

  float m = -INFINITY;   // running row max   (row = lane & 15, replicated)
  float l = 0.0f;        // running row sum
  const v8f vzero = {0.f, 0.f, 0.f, 0.f, 0.f, 0.f, 0.f, 0.f};
  v8f acc0 = vzero, acc1 = vzero, acc2 = vzero, acc3 = vzero;

  // pipelined adjacency + e_dst tile registers
  int4 ma0, ma1, mb0, mb1;
  float4 da0, da1, db0, db1;
  auto issueME = [&](int j0, int4& a0, int4& a1, int4& a2, int4& a3,
                     float4& d0, float4& d1, float4& d2, float4& d3) {
    const int ka = j0 + 8 * half;   // K = t + 8*half       (t = 0..7)
    const int kb = ka + 16;         // K = t + 8*half + 8   (t = 8..15)
    a0 = *(const int4*)(adjp + ka);
    a1 = *(const int4*)(adjp + ka + 4);
    a2 = *(const int4*)(adjp + kb);
    a3 = *(const int4*)(adjp + kb + 4);
    d0 = *(const float4*)(edp + ka);
    d1 = *(const float4*)(edp + ka + 4);
    d2 = *(const float4*)(edp + kb);
    d3 = *(const float4*)(edp + kb + 4);
  };
  issueME(0, ma0, ma1, mb0, mb1, da0, da1, db0, db1);

  for (int j0 = 0; j0 < kN; j0 += 32) {
    // ---- issue next tile's mask/e_dst loads (clamped; last one redundant)
    const int jn = (j0 + 32 < kN) ? (j0 + 32) : j0;
    int4 na0, na1, nb0, nb1;
    float4 ea0, ea1, eb0, eb1;
    issueME(jn, na0, na1, nb0, nb1, ea0, ea1, eb0, eb1);

    // ---- issue this tile's Wh B-fragments (latency covered by VALU below)
    const bf16_t* wcol = whp + (size_t)j0 * (size_t)O + mrow;
    v16bf b0, b1, b2, b3;
#pragma unroll
    for (int s = 0; s < 16; ++s) {
      const int kk = (s < 8) ? (8 * half + s) : (16 + 8 * half + (s - 8));
      const bf16_t* wp = wcol + (size_t)kk * O;
      b0[s] = wp[0]; b1[s] = wp[16]; b2[s] = wp[32]; b3[s] = wp[48];
    }
    __builtin_prefetch((const void*)(whp + (size_t)(j0 + 32 + lane) * O), 0, 0);

    const int msk[16] = {ma0.x, ma0.y, ma0.z, ma0.w, ma1.x, ma1.y, ma1.z,
                         ma1.w, mb0.x, mb0.y, mb0.z, mb0.w, mb1.x, mb1.y,
                         mb1.z, mb1.w};
    const float ev[16] = {da0.x, da0.y, da0.z, da0.w, da1.x, da1.y, da1.z,
                          da1.w, db0.x, db0.y, db0.z, db0.w, db1.x, db1.y,
                          db1.z, db1.w};

    float s[16];
    float tmax = kNeg;
#pragma unroll
    for (int t = 0; t < 16; ++t) {
      float x = eSrc + ev[t];
      x = x > 0.f ? x : kAlpha * x;          // leaky_relu
      s[t] = (msk[t] > 0) ? x : kNeg;        // adjacency mask
      tmax = fmaxf(tmax, s[t]);
    }
    tmax = fmaxf(tmax, __shfl_xor(tmax, 16, 32));  // combine K halves per row

    const float mnew = fmaxf(m, tmax);
    const float scale = __expf(m - mnew);
    m = mnew;

    float p[16];
    float psum = 0.f;
#pragma unroll
    for (int t = 0; t < 16; ++t) {
      p[t] = __expf(s[t] - mnew);
      psum += p[t];
    }
    psum += __shfl_xor(psum, 16, 32);
    l = l * scale + psum;

    // rescale accumulators: C element t lives in row (t + 8*half)
#pragma unroll
    for (int t = 0; t < 8; ++t) {
      const float sc = __shfl(scale, t + 8 * half, 32);
      acc0[t] *= sc; acc1[t] *= sc; acc2[t] *= sc; acc3[t] *= sc;
    }

    v16bf af;
#pragma unroll
    for (int t = 0; t < 16; ++t) af[t] = (bf16_t)p[t];

    acc0 = wmma_bf16(af, b0, acc0);
    acc1 = wmma_bf16(af, b1, acc1);
    acc2 = wmma_bf16(af, b2, acc2);
    acc3 = wmma_bf16(af, b3, acc3);

    // rotate pipelined tile
    ma0 = na0; ma1 = na1; mb0 = nb0; mb1 = nb1;
    da0 = ea0; da1 = ea1; db0 = eb0; db1 = eb1;
  }

  // ---- epilogue: normalize by row sum, ELU, store
  const int outStride = HH * O;
#pragma unroll
  for (int t = 0; t < 8; ++t) {
    const int r = t + 8 * half;
    const float linv = 1.0f / __shfl(l, r, 32);
    const size_t base = ((size_t)b * kN + row0 + r) * (size_t)outStride +
                        (size_t)hh * O + colbase + mrow;
    const float v0 = elu1(acc0[t] * linv);
    const float v1 = elu1(acc1[t] * linv);
    const float v2 = elu1(acc2[t] * linv);
    const float v3 = elu1(acc3[t] * linv);
    if (OUT_BF16) {
      bf16_t* op = (bf16_t*)outp;
      op[base +  0] = (bf16_t)v0;
      op[base + 16] = (bf16_t)v1;
      op[base + 32] = (bf16_t)v2;
      op[base + 48] = (bf16_t)v3;
    } else {
      float* op = (float*)outp;
      op[base +  0] = v0;
      op[base + 16] = v1;
      op[base + 32] = v2;
      op[base + 48] = v3;
    }
  }
}

// ---------------------------------------------------------------------------
extern "C" void kernel_launch(void* const* d_in, const int* in_sizes, int n_in,
                              void* d_out, int out_size, void* d_ws, size_t ws_size,
                              hipStream_t stream) {
  const float* features = (const float*)d_in[0];  // [8,1024,256]
  const int*   adjs     = (const int*)d_in[1];    // [8,1024,1024]
  const float* W1       = (const float*)d_in[2];  // [8,256,64]
  const float* a1       = (const float*)d_in[3];  // [8,128]
  const float* W2       = (const float*)d_in[4];  // [1,512,256]
  const float* a2       = (const float*)d_in[5];  // [1,512]

  char* ws = (char*)d_ws;
  size_t off = 0;
  auto alloc = [&](size_t bytes) -> void* {
    void* p = ws + off;
    off = (off + bytes + 255) & ~(size_t)255;
    return p;
  };
  bf16_t* Wh1  = (bf16_t*)alloc((size_t)kB * 8 * kN * 64 * 2);  // 8 MB
  float*  es1  = (float*)alloc((size_t)kB * 8 * kN * 4);
  float*  ed1  = (float*)alloc((size_t)kB * 8 * kN * 4);
  bf16_t* h1   = (bf16_t*)alloc((size_t)kB * kN * 512 * 2);     // 8 MB
  bf16_t* Wh2  = (bf16_t*)alloc((size_t)kB * kN * 256 * 2);     // 4 MB
  float*  es2  = (float*)alloc((size_t)kB * kN * 4);
  float*  ed2  = (float*)alloc((size_t)kB * kN * 4);
  bf16_t* W1bf = (bf16_t*)alloc((size_t)8 * 256 * 64 * 2);      // 256 KB
  bf16_t* W2bf = (bf16_t*)alloc((size_t)512 * 256 * 2);         // 256 KB

  // ---- one-time weight conversion to bf16
  cvt_bf16_kernel<<<(8 * 256 * 64) / 256, 256, 0, stream>>>(W1, W1bf,
                                                            8 * 256 * 64);
  cvt_bf16_kernel<<<(512 * 256) / 256, 256, 0, stream>>>(W2, W2bf, 512 * 256);

  // ---- Layer 1: Wh1 = X @ W1  (waves = 8*8*64 = 4096, 4 waves/block)
  gemm_wh_kernel<false><<<1024, 128, 0, stream>>>(
      features, W1bf, Wh1, /*Kdim=*/256, /*O=*/64, /*HH=*/8, /*colChunks=*/1,
      4096);
  edot_kernel<<<(kB * 8 * kN) / 256, 256, 0, stream>>>(
      Wh1, a1, es1, ed1, /*HH=*/8, /*O=*/64, kB * 8 * kN);
  flash_gat_kernel<true><<<1024, 128, 0, stream>>>(
      es1, ed1, Wh1, adjs, h1, /*HH=*/8, /*O=*/64, /*colChunks=*/1, 4096);

  // ---- Layer 2: Wh2 = h1 @ W2  (waves = 8*64*4 = 2048)
  gemm_wh_kernel<true><<<512, 128, 0, stream>>>(
      h1, W2bf, Wh2, /*Kdim=*/512, /*O=*/256, /*HH=*/1, /*colChunks=*/4, 2048);
  edot_kernel<<<(kB * kN) / 256, 256, 0, stream>>>(
      Wh2, a2, es2, ed2, /*HH=*/1, /*O=*/256, kB * kN);
  flash_gat_kernel<false><<<512, 128, 0, stream>>>(
      es2, ed2, Wh2, adjs, d_out, /*HH=*/1, /*O=*/256, /*colChunks=*/4, 2048);
}
